// KAFINBlock_4260607557781
// MI455X (gfx1250) — matrix operand order repaired and checked by reading, not compiled, not run on hardware
//
#include <hip/hip_runtime.h>

typedef __attribute__((ext_vector_type(16))) _Float16     v16h;
typedef __attribute__((ext_vector_type(8)))  float        v8f;
typedef __attribute__((ext_vector_type(4)))  unsigned int u32x4;
typedef __attribute__((ext_vector_type(8)))  int          i32x8;
typedef __attribute__((ext_vector_type(4)))  int          i32x4;

#define B_TOTAL 16384
#define D_FEAT  64
#define H_DIM   256
#define ROWS_PER_BLOCK 128   // 8 waves * 16 rows

// ---------------------------------------------------------------------------
// Workspace layout:
//   [0      , 8 MiB)  : W2 as f16, pre-swizzled into WMMA B-fragment order
//   [8 MiB  , 12 MiB) : h_cat fp32 [B][D]
// ---------------------------------------------------------------------------
#define WS_HCAT_OFF ((size_t)8 * 1024 * 1024)

#if defined(__HIP_DEVICE_COMPILE__) && __has_builtin(__builtin_amdgcn_tensor_load_to_lds)
#define USE_TDM 1
#else
#define USE_TDM 0
#endif

// ---------------------------------------------------------------------------
// Kernel 1: pack W2[d][k][n] (fp32) into f16 WMMA B fragments.
// Fragment element order:  (((d*16 + ntile)*8 + kchunk)*32 + lane)*16 + e
// B order used here (contiguous k, 32x16): lanes 0-15 hold n=lane, k = kbase + e (e=0..15);
//                           lanes 16-31 hold n=lane-16, k = kbase + 16 + e.
// ---------------------------------------------------------------------------
__global__ __launch_bounds__(256) void pack_w2_kernel(
    const float* __restrict__ W2, _Float16* __restrict__ wsB) {
  const int idx  = blockIdx.x * 256 + threadIdx.x;   // 0 .. 262143
  const int lane = idx & 31;
  const int c    = (idx >> 5) & 7;    // k-chunk
  const int nt   = (idx >> 8) & 15;   // n-tile
  const int d    = idx >> 12;         // feature
  const int kbase = c * 32 + ((lane & 16) ? 16 : 0);
  const int n     = nt * 16 + (lane & 15);
  const float* __restrict__ w2d = W2 + (size_t)d * (H_DIM * H_DIM);

  union { _Float16 h[16]; uint4 u[2]; } pk;
#pragma unroll
  for (int e = 0; e < 16; ++e)
    pk.h[e] = (_Float16)w2d[(size_t)(kbase + e) * H_DIM + n];

  uint4* dst = (uint4*)(wsB + (size_t)idx * 16);
  dst[0] = pk.u[0];
  dst[1] = pk.u[1];
}

// ---------------------------------------------------------------------------
// TDM: DMA one contiguous 64 KiB block (8192 x 8-byte elements) into LDS.
// Descriptor per CDNA5 ISA (group0 + group1, 1-row tile).
// ---------------------------------------------------------------------------
#if USE_TDM
__device__ __forceinline__ void tdm_copy_64k(unsigned int lds_addr,
                                             unsigned long long gaddr) {
  u32x4 g0 = {};
  g0[0] = 1u;                                   // count=1, user descriptor
  g0[1] = lds_addr;                             // lds_addr (bytes)
  g0[2] = (unsigned int)gaddr;                  // global_addr[31:0]
  g0[3] = (unsigned int)((gaddr >> 32) & 0x1FFFFFFull) | (2u << 30); // +type=2

  i32x8 g1 = {};
  g1[0] = (int)(3u << 16);          // workgroup_mask=0, data_size=3 (8 bytes)
  g1[1] = (int)(8192u << 16);       // tensor_dim0[15:0]=8192 @ bits 63:48
  g1[2] = (int)(1u << 16);          // tensor_dim0[31:16]=0; tensor_dim1=1
  g1[3] = (int)(8192u << 16);       // tile_dim0=8192 @ bits 127:112
  g1[4] = 0;                        // tile_dim1=0 (unused), tile_dim2=0
  g1[5] = 8192;                     // tensor_dim0_stride
  g1[6] = 0;
  g1[7] = 0;

  i32x4 z4 = {};
  i32x8 z8 = {};
#if defined(__clang_major__) && __clang_major__ >= 23
  __builtin_amdgcn_tensor_load_to_lds(g0, g1, z4, z4, z8, 0);   // 6-arg form
#else
  __builtin_amdgcn_tensor_load_to_lds(g0, g1, z4, z4, 0);       // 5-arg form
#endif
}
#endif

// ---------------------------------------------------------------------------
// 8 n-tiles x 8 k-chunks of v_wmma_f32_16x16x32_f16 with fused b2/relu/W3.
// ---------------------------------------------------------------------------
__device__ __forceinline__ void mlp_tiles(const v16h* __restrict__ Bfrag,
                                          const v16h A[8], int lane, int m,
                                          const float* __restrict__ b2s,
                                          const float* __restrict__ W3s,
                                          float hacc[8], int ntBase) {
#pragma unroll
  for (int nti = 0; nti < 8; ++nti) {
    const int nt = ntBase + nti;
    v8f acc = {};
#pragma unroll
    for (int c = 0; c < 8; ++c) {
      const v16h b = Bfrag[(nt * 8 + c) * 32 + lane];
      acc = __builtin_amdgcn_wmma_f32_16x16x32_f16(
          /*neg_a=*/false, A[c], /*neg_b=*/false, b,
          /*c_mod=*/(short)0, acc, /*reuse_a=*/false, /*reuse_b=*/false);
    }
    const int   n   = nt * 16 + m;      // this lane's output column
    const float b2v = b2s[n];
    const float w3v = W3s[n];
#pragma unroll
    for (int r = 0; r < 8; ++r) {       // acc[r] -> row (hi?8:0)+r, col n
      const float v = fmaxf(acc[r] + b2v, 0.f);   // a2 = relu(. + b2)
      hacc[r] = fmaf(v, w3v, hacc[r]);            // partial dot with W3
    }
  }
}

// ---------------------------------------------------------------------------
// Kernel 2: per-feature 3-layer MLP -> h_cat[b][d], WMMA f16 with f32 accum.
// grid = (B/128, D).  256 threads = 8 waves; wave w owns rows [w*16, w*16+16).
// ---------------------------------------------------------------------------
__global__ __launch_bounds__(256) void feature_mlp_kernel(
    const float* __restrict__ x,  const float* __restrict__ W1,
    const float* __restrict__ b1, const float* __restrict__ b2,
    const float* __restrict__ W3, const float* __restrict__ b3,
    const _Float16* __restrict__ wsB, float* __restrict__ hcat) {
  __shared__ v16h  Bfrag[16 * 8 * 32];   // 128 KiB: full W2_d in fragment order
  __shared__ float W1s[H_DIM], b1s[H_DIM], b2s[H_DIM], W3s[H_DIM];

  const int tid  = threadIdx.x;
  const int lane = tid & 31;
  const int wave = tid >> 5;
  const int m    = lane & 15;          // row within the wave's 16-row tile
  const int hi   = (lane >> 4) & 1;    // lane half (selects K sub-range)
  const int d    = blockIdx.y;
  const int rowbase = blockIdx.x * ROWS_PER_BLOCK;

  // --- Stage B fragments: TDM double-buffer (halves), fallback = vector copy
#if USE_TDM
  if (wave == 0) {
    const unsigned int      lds0 = (unsigned int)(uintptr_t)(&Bfrag[0]);
    const unsigned long long g   =
        (unsigned long long)(uintptr_t)(wsB + (size_t)d * 65536);
    tdm_copy_64k(lds0,          g);            // n-tiles 0..7
    tdm_copy_64k(lds0 + 65536u, g + 65536ull); // n-tiles 8..15
  }
#else
  {
    const uint4* __restrict__ src = (const uint4*)(wsB + (size_t)d * 65536);
    uint4* dst = (uint4*)Bfrag;        // 8192 uint4 total
#pragma unroll 4
    for (int i = 0; i < 32; ++i)
      dst[tid + i * 256] = src[tid + i * 256];
  }
#endif
  W1s[tid] = W1[d * H_DIM + tid];
  b1s[tid] = b1[d * H_DIM + tid];
  b2s[tid] = b2[d * H_DIM + tid];
  W3s[tid] = W3[d * H_DIM + tid];
#if USE_TDM
  if (wave == 0) __builtin_amdgcn_s_wait_tensorcnt(1);  // first half landed
#endif
  __syncthreads();

  // --- Build A fragments in registers: a1 = relu(x*W1 + b1) -> f16 ----------
  // A order used here (contiguous k, as B): lanes 0-15: k=kb+e (e=0..15);
  //                           lanes 16-31: shift K by +16.
  const int   row = rowbase + wave * 16 + m;
  const float xv  = x[(size_t)row * D_FEAT + d];
  v16h A[8];
#pragma unroll
  for (int c = 0; c < 8; ++c) {
    const int kb = c * 32;
#pragma unroll
    for (int e = 0; e < 16; ++e) {
      const int k = kb + e + (hi ? 16 : 0);
      const float a = fmaxf(fmaf(xv, W1s[k], b1s[k]), 0.f);
      A[c][e] = (_Float16)a;
    }
  }

  // --- Compute: first half overlaps with the second TDM transfer ------------
  float hacc[8] = {0.f, 0.f, 0.f, 0.f, 0.f, 0.f, 0.f, 0.f};
  mlp_tiles(Bfrag, A, lane, m, b2s, W3s, hacc, 0);
#if USE_TDM
  if (wave == 0) __builtin_amdgcn_s_wait_tensorcnt(0);  // second half landed
  __syncthreads();
#endif
  mlp_tiles(Bfrag, A, lane, m, b2s, W3s, hacc, 8);

  // --- Reduce partials across the 16 lanes that share a row; + b3; store ----
  const float b3v = b3[d];
#pragma unroll
  for (int r = 0; r < 8; ++r) {
    float v = hacc[r];
    v += __shfl_xor(v, 1);
    v += __shfl_xor(v, 2);
    v += __shfl_xor(v, 4);
    v += __shfl_xor(v, 8);
    if (m == 0) {
      const int rrow = rowbase + wave * 16 + (hi ? 8 : 0) + r;
      hcat[(size_t)rrow * D_FEAT + d] = v + b3v;
    }
  }
}

// ---------------------------------------------------------------------------
// Kernel 3: attention fusion + cross MLP + g_func.  One block per batch row.
// ---------------------------------------------------------------------------
__global__ __launch_bounds__(256) void fuse_kernel(
    const float* __restrict__ hcat,
    const float* __restrict__ Wq,  const float* __restrict__ bq,
    const float* __restrict__ Wk,  const float* __restrict__ bk,
    const float* __restrict__ Ws,  const float* __restrict__ bs,
    const float* __restrict__ Wc1, const float* __restrict__ bc1,
    const float* __restrict__ Wc2, const float* __restrict__ bc2,
    const float* __restrict__ Wg1, const float* __restrict__ bg1,
    const float* __restrict__ Wg2, const float* __restrict__ bg2,
    float* __restrict__ out) {
  __shared__ float hs[D_FEAT];
  __shared__ float ts[H_DIM];
  __shared__ float red[H_DIM];
  __shared__ float ss[D_FEAT];

  const int tid = threadIdx.x;
  const int b   = blockIdx.x;

  if (tid < D_FEAT) hs[tid] = hcat[(size_t)b * D_FEAT + tid];
  __syncthreads();

  // thread j computes q[j], k[j], cross-hidden[j]
  float qa = bq[tid], ka = bk[tid], c1 = bc1[tid];
  const float* __restrict__ wqr = Wq  + tid * D_FEAT;
  const float* __restrict__ wkr = Wk  + tid * D_FEAT;
  const float* __restrict__ wcr = Wc1 + tid * D_FEAT;
#pragma unroll 8
  for (int dd = 0; dd < D_FEAT; ++dd) {
    const float hv = hs[dd];
    qa = fmaf(hv, wqr[dd], qa);
    ka = fmaf(hv, wkr[dd], ka);
    c1 = fmaf(hv, wcr[dd], c1);
  }
  ts[tid]  = tanhf(qa * ka);
  red[tid] = fmaxf(c1, 0.f) * Wc2[tid];
  __syncthreads();

  // tree-reduce cross term over 256 threads
  for (int st = 128; st > 0; st >>= 1) {
    if (tid < st) red[tid] += red[tid + st];
    __syncthreads();
  }

  // scores[d] = tanh(q*k) . Ws[d,:] + bs[d]
  if (tid < D_FEAT) {
    float sv = bs[tid];
    const float* __restrict__ wsr = Ws + tid * H_DIM;
#pragma unroll 8
    for (int j = 0; j < H_DIM; ++j) sv = fmaf(ts[j], wsr[j], sv);
    ss[tid] = sv;
  }
  __syncthreads();

  if (tid == 0) {
    float mx = -1e30f;
    for (int dd = 0; dd < D_FEAT; ++dd) mx = fmaxf(mx, ss[dd]);
    float sum = 0.f, wsum = 0.f;
    for (int dd = 0; dd < D_FEAT; ++dd) {
      const float e = __expf(ss[dd] - mx);
      sum += e;
      wsum = fmaf(e, hs[dd], wsum);
    }
    const float comb = wsum / sum + red[0] + bc2[0];
    float o = bg2[0];
    for (int hh = 0; hh < H_DIM; ++hh)
      o = fmaf(fmaxf(fmaf(comb, Wg1[hh], bg1[hh]), 0.f), Wg2[hh], o);
    out[b] = o;
  }
}

// ---------------------------------------------------------------------------
extern "C" void kernel_launch(void* const* d_in, const int* in_sizes, int n_in,
                              void* d_out, int out_size, void* d_ws, size_t ws_size,
                              hipStream_t stream) {
  const float* x   = (const float*)d_in[0];
  const float* W1  = (const float*)d_in[1];
  const float* b1  = (const float*)d_in[2];
  const float* W2  = (const float*)d_in[3];
  const float* b2  = (const float*)d_in[4];
  const float* W3  = (const float*)d_in[5];
  const float* b3  = (const float*)d_in[6];
  const float* Wq  = (const float*)d_in[7];
  const float* bq  = (const float*)d_in[8];
  const float* Wk  = (const float*)d_in[9];
  const float* bk  = (const float*)d_in[10];
  const float* Ws  = (const float*)d_in[11];
  const float* bs  = (const float*)d_in[12];
  const float* Wc1 = (const float*)d_in[13];
  const float* bc1 = (const float*)d_in[14];
  const float* Wc2 = (const float*)d_in[15];
  const float* bc2 = (const float*)d_in[16];
  const float* Wg1 = (const float*)d_in[17];
  const float* bg1 = (const float*)d_in[18];
  const float* Wg2 = (const float*)d_in[19];
  const float* bg2 = (const float*)d_in[20];
  float* out = (float*)d_out;

  _Float16* wsB  = (_Float16*)d_ws;
  float*    hcat = (float*)((char*)d_ws + WS_HCAT_OFF);

  // 1) W2 -> f16 WMMA-fragment order (8 MiB, stays hot in 192 MB L2)
  pack_w2_kernel<<<1024, 256, 0, stream>>>(W2, wsB);

  // 2) dominant batched GEMM with fused a1-build and W3/b2/b3 epilogue
  dim3 gridB(B_TOTAL / ROWS_PER_BLOCK, D_FEAT);
  feature_mlp_kernel<<<gridB, 256, 0, stream>>>(x, W1, b1, b2, W3, b3, wsB, hcat);

  // 3) cheap fusion tail
  fuse_kernel<<<B_TOTAL, 256, 0, stream>>>(hcat, Wq, bq, Wk, bk, Ws, bs,
                                           Wc1, bc1, Wc2, bc2, Wg1, bg1,
                                           Wg2, bg2, out);
}
